// MiTransformer_25254407700653
// MI455X (gfx1250) — matrix operand order repaired and checked by reading, not compiled, not run on hardware
//
#include <hip/hip_runtime.h>
#include <math.h>

#define DEVINL __device__ __forceinline__

typedef __bf16 bf16_t;
typedef __bf16 v16bf __attribute__((ext_vector_type(16)));
typedef float  v8f   __attribute__((ext_vector_type(8)));
typedef float  f32x4 __attribute__((ext_vector_type(4)));
typedef int    v4i   __attribute__((ext_vector_type(4)));

constexpr int kB  = 16;
constexpr int kNV = 512;
constexpr int kSL = 1024;
constexpr int kD  = 512;
constexpr int kH  = 16;
constexpr int kDH = 32;
constexpr int kE  = 8;
constexpr int kFF = 16;
constexpr int kNL = 2;
constexpr int kNI = 64;
constexpr int kT  = kB * kNV;

#if defined(__AMDGCN__) && __has_builtin(__builtin_amdgcn_global_load_async_to_lds_b128) && __has_builtin(__builtin_amdgcn_s_wait_asynccnt)
#define USE_ASYNC_COPY 1
#else
#define USE_ASYNC_COPY 0
#endif

DEVINL void lds_copy16(bf16_t* dst, const bf16_t* src) {
#if USE_ASYNC_COPY
  __builtin_amdgcn_global_load_async_to_lds_b128(
      (__attribute__((address_space(1))) v4i*)(src),
      (__attribute__((address_space(3))) v4i*)(dst), 0, 0);
  __builtin_amdgcn_global_load_async_to_lds_b128(
      (__attribute__((address_space(1))) v4i*)(src + 8),
      (__attribute__((address_space(3))) v4i*)(dst + 8), 0, 0);
#else
  *(f32x4*)dst       = *(const f32x4*)src;
  *(f32x4*)(dst + 8) = *(const f32x4*)(src + 8);
#endif
}

DEVINL void lds_zero16(bf16_t* dst) {
  f32x4 z = {0.f, 0.f, 0.f, 0.f};
  *(f32x4*)dst = z; *(f32x4*)(dst + 8) = z;
}

DEVINL void async_fence() {
#if USE_ASYNC_COPY
  __builtin_amdgcn_s_wait_asynccnt(0);
#endif
}

// ---------------------------------------------------------------------------
// Generic batched WMMA GEMM:  C[b] = alpha * A[b] @ B[b]^T (+ bias)
// A: [M,K] rows K-contiguous (lda), B: [N,K] rows K-contiguous (ldb).
// Block tile 128x128, 8 waves (2x4), wave tile 64x32, K-step 32, double-buffered LDS.
// ---------------------------------------------------------------------------
constexpr int BM = 128, BN = 128, BK = 32;

__global__ __launch_bounds__(256)
void wmma_gemm_kernel(const bf16_t* __restrict__ Amat, long aBatch, int lda,
                      const bf16_t* __restrict__ Bmat, long bBatch, int ldb,
                      float* __restrict__ Cmat, long cBatch, int ldc,
                      bf16_t* __restrict__ Cbf, long cbfBatch,
                      const float* __restrict__ bias, float alpha,
                      int M, int N, int K) {
  __shared__ __align__(16) bf16_t As[2][BM][BK];
  __shared__ __align__(16) bf16_t Bs[2][BN][BK];

  const int bz = blockIdx.z;
  Amat += (long)bz * aBatch;
  Bmat += (long)bz * bBatch;
  Cmat += (long)bz * cBatch;
  if (Cbf) Cbf += (long)bz * cbfBatch;

  const int m0 = blockIdx.y * BM;
  const int n0 = blockIdx.x * BN;
  const int tid  = threadIdx.x;
  const int lane = tid & 31;
  const int wave = tid >> 5;
  const int wRow = wave >> 2;   // 0..1 -> 64 rows
  const int wCol = wave & 3;    // 0..3 -> 32 cols

  const int ldRow = tid >> 1;         // 0..127
  const int ldCol = (tid & 1) * 16;   // 0 or 16

  auto fill_tile = [&](int buf, int kt) {
    const int k0 = kt * BK;
    {
      bf16_t* dst = &As[buf][ldRow][ldCol];
      if (m0 + ldRow < M) lds_copy16(dst, Amat + (long)(m0 + ldRow) * lda + k0 + ldCol);
      else                lds_zero16(dst);
    }
    {
      bf16_t* dst = &Bs[buf][ldRow][ldCol];
      if (n0 + ldRow < N) lds_copy16(dst, Bmat + (long)(n0 + ldRow) * ldb + k0 + ldCol);
      else                lds_zero16(dst);
    }
  };

  const v8f zero8 = {0.f,0.f,0.f,0.f,0.f,0.f,0.f,0.f};
  v8f acc[4][2];
#pragma unroll
  for (int mi = 0; mi < 4; ++mi)
#pragma unroll
    for (int ni = 0; ni < 2; ++ni) acc[mi][ni] = zero8;

  const int kTiles = K / BK;
  fill_tile(0, 0);
  async_fence();
  __syncthreads();

  const int fr = lane & 15;   // row within 16 (A) / col within 16 (B)
  const int fh = lane >> 4;   // lane half selects K grouping

  for (int kt = 0; kt < kTiles; ++kt) {
    const int cur = kt & 1, nxt = cur ^ 1;
    if (kt + 1 < kTiles) fill_tile(nxt, kt + 1);

    union { v16bf v; f32x4 q[2]; } bfr[2];
#pragma unroll
    for (int ni = 0; ni < 2; ++ni) {
      // B 32x16 layout: lane half 0: K 0..7 & 16..23 ; lane half 1: K 8..15 & 24..31
      const bf16_t* bp = &Bs[cur][wCol * 32 + ni * 16 + fr][fh * 8];
      bfr[ni].q[0] = *(const f32x4*)bp;
      bfr[ni].q[1] = *(const f32x4*)(bp + 16);
    }
#pragma unroll
    for (int mi = 0; mi < 4; ++mi) {
      union { v16bf v; f32x4 q[2]; } afr;
      // A 16x32 layout: lane half 0: K 0..7 & 16..23 ; lane half 1: K 8..15 & 24..31
      const bf16_t* ap = &As[cur][wRow * 64 + mi * 16 + fr][fh * 8];
      afr.q[0] = *(const f32x4*)ap;
      afr.q[1] = *(const f32x4*)(ap + 16);
#pragma unroll
      for (int ni = 0; ni < 2; ++ni) {
        acc[mi][ni] = __builtin_amdgcn_wmma_f32_16x16x32_bf16(
            false, afr.v, false, bfr[ni].v, (short)0, acc[mi][ni], false, false);
      }
    }
    async_fence();
    __syncthreads();
  }

  const int cCol  = lane & 15;
  const int rHalf = (lane >> 4) * 8;
#pragma unroll
  for (int mi = 0; mi < 4; ++mi)
#pragma unroll
    for (int ni = 0; ni < 2; ++ni)
#pragma unroll
      for (int e = 0; e < 8; ++e) {
        const int row = m0 + wRow * 64 + mi * 16 + rHalf + e;
        const int col = n0 + wCol * 32 + ni * 16 + cCol;
        if (row < M && col < N) {
          float v = acc[mi][ni][e] * alpha;
          if (bias) v += bias[col];
          Cmat[(long)row * ldc + col] = v;
          if (Cbf) Cbf[(long)row * ldc + col] = (bf16_t)v;
        }
      }
}

// ---------------------------------------------------------------------------
// RevIN: per (b,n) mean/std over time, write normalized series as bf16 rows.
// ---------------------------------------------------------------------------
__global__ __launch_bounds__(256)
void revin_kernel(const float* __restrict__ xseq, const float* __restrict__ rev_w,
                  const float* __restrict__ rev_b, float* __restrict__ meanB,
                  float* __restrict__ stdB, bf16_t* __restrict__ xn) {
  const int bn = blockIdx.x;
  const int b = bn >> 9, n = bn & (kNV - 1);
  const float* xp = xseq + (long)b * kSL * kNV + n;   // stride kNV over time
  const int tid = threadIdx.x;
  float s = 0.f, s2 = 0.f;
  for (int l = tid; l < kSL; l += 256) { float v = xp[(long)l * kNV]; s += v; s2 += v * v; }
  __shared__ float r1[256], r2[256];
  r1[tid] = s; r2[tid] = s2; __syncthreads();
  for (int o = 128; o; o >>= 1) { if (tid < o) { r1[tid] += r1[tid + o]; r2[tid] += r2[tid + o]; } __syncthreads(); }
  const float m   = r1[0] / kSL;
  const float var = r2[0] / kSL - m * m;
  const float sd  = sqrtf(var + 1e-5f);
  const float w = rev_w[n], bb = rev_b[n], inv = 1.f / sd;
  for (int l = tid; l < kSL; l += 256) {
    float v = xp[(long)l * kNV];
    xn[(long)bn * kSL + l] = (bf16_t)((v - m) * inv * w + bb);
  }
  if (tid == 0) { meanB[bn] = m; stdB[bn] = sd; }
}

__global__ void cast_bf16_kernel(const float* __restrict__ in, bf16_t* __restrict__ out, long n) {
  long i = (long)blockIdx.x * 256 + threadIdx.x;
  if (i < n) out[i] = (bf16_t)in[i];
}

// v [B,NV,D] f32 -> vT [B,H,dh,NV] bf16
__global__ void transpose_v_kernel(const float* __restrict__ v, bf16_t* __restrict__ vT) {
  long i = (long)blockIdx.x * 256 + threadIdx.x;
  if (i >= (long)kT * kD) return;
  const int  d  = (int)(i & (kD - 1));
  const long bn = i >> 9;
  const int  n  = (int)(bn & (kNV - 1));
  const int  b  = (int)(bn >> 9);
  const int  h  = d >> 5, j = d & 31;
  vT[(((long)b * kH + h) * kDH + j) * kNV + n] = (bf16_t)v[i];
}

// row softmax over 512 cols, output bf16
__global__ __launch_bounds__(256)
void softmax_kernel(const float* __restrict__ S, bf16_t* __restrict__ P) {
  const long row = blockIdx.x;
  const float* sr = S + row * (long)kNV;
  const int tid = threadIdx.x;
  const float a = sr[tid], c = sr[tid + 256];
  __shared__ float red[256];
  red[tid] = fmaxf(a, c); __syncthreads();
  for (int o = 128; o; o >>= 1) { if (tid < o) red[tid] = fmaxf(red[tid], red[tid + o]); __syncthreads(); }
  const float mx = red[0]; __syncthreads();
  const float e0 = expf(a - mx), e1 = expf(c - mx);
  red[tid] = e0 + e1; __syncthreads();
  for (int o = 128; o; o >>= 1) { if (tid < o) red[tid] += red[tid + o]; __syncthreads(); }
  const float inv = 1.f / red[0];
  P[row * kNV + tid]       = (bf16_t)(e0 * inv);
  P[row * kNV + tid + 256] = (bf16_t)(e1 * inv);
}

// out = LayerNorm(X + Y) * g + b ; optional bf16 mirror
__global__ __launch_bounds__(256)
void add_ln_kernel(const float* __restrict__ X, const float* __restrict__ Y,
                   const float* __restrict__ g, const float* __restrict__ bvec,
                   float* __restrict__ out, bf16_t* __restrict__ outbf) {
  const int row = blockIdx.x, tid = threadIdx.x;
  const long base = (long)row * kD;
  float t0 = X[base + tid]       + (Y ? Y[base + tid]       : 0.f);
  float t1 = X[base + tid + 256] + (Y ? Y[base + tid + 256] : 0.f);
  __shared__ float r1[256], r2[256];
  r1[tid] = t0 + t1; r2[tid] = t0 * t0 + t1 * t1; __syncthreads();
  for (int o = 128; o; o >>= 1) { if (tid < o) { r1[tid] += r1[tid + o]; r2[tid] += r2[tid + o]; } __syncthreads(); }
  const float m   = r1[0] / kD;
  const float var = r2[0] / kD - m * m;
  const float inv = rsqrtf(var + 1e-5f);
  const float o0 = (t0 - m) * inv * g[tid]       + bvec[tid];
  const float o1 = (t1 - m) * inv * g[tid + 256] + bvec[tid + 256];
  out[base + tid] = o0; out[base + tid + 256] = o1;
  if (outbf) { outbf[base + tid] = (bf16_t)o0; outbf[base + tid + 256] = (bf16_t)o1; }
}

// instruction-token mean over NI
__global__ void ins_mean_kernel(const float* __restrict__ ins, float* __restrict__ out) {
  int d = blockIdx.x * 256 + threadIdx.x;
  if (d >= kD) return;
  float s = 0.f;
  for (int n = 0; n < kNI; ++n) s += ins[(long)n * kD + d];
  out[d] = s / kNI;
}

// gam[e] = insMean . gw[e]; bet[e*D+f] = insMean . bw[(e*D+f)*D ...]
__global__ void eilm_kernel(const float* __restrict__ insMean, const float* __restrict__ gw,
                            const float* __restrict__ bw, float* __restrict__ gam,
                            float* __restrict__ bet) {
  int idx = blockIdx.x * 256 + threadIdx.x;
  if (idx < kE) {
    float s = 0.f; const float* p = gw + (long)idx * kD;
    for (int d = 0; d < kD; ++d) s += insMean[d] * p[d];
    gam[idx] = s;
  }
  if (idx < kE * kD) {
    float s = 0.f; const float* p = bw + (long)idx * kD;
    for (int d = 0; d < kD; ++d) s += insMean[d] * p[d];
    bet[idx] = s;
  }
}

// MoE: softmax gate, top-2 (no renorm), SwiGLU experts (FF=16), EiLM mod, combine.
__global__ __launch_bounds__(256)
void moe_kernel(const float* __restrict__ x1, const float* __restrict__ gateW,
                const float* __restrict__ Wg, const float* __restrict__ Wu,
                const float* __restrict__ Wd, const float* __restrict__ gam,
                const float* __restrict__ bet, float* __restrict__ moe_out) {
  const int t = blockIdx.x, tid = threadIdx.x;
  __shared__ float tok[kD];
  __shared__ float logits[kE];
  __shared__ float wcomb[kE];
  __shared__ float hbuf[kFF];

  for (int c = tid; c < kD; c += 256) tok[c] = x1[(long)t * kD + c];
  __syncthreads();

  const int wv = tid >> 5, lnid = tid & 31;
  {
    float a = 0.f;
    const float* gw = gateW + (long)wv * kD;
    for (int d = lnid; d < kD; d += 32) a += tok[d] * gw[d];
    for (int o = 16; o; o >>= 1) a += __shfl_down(a, o, 32);
    if (lnid == 0) logits[wv] = a;
  }
  __syncthreads();
  if (tid == 0) {
    float mx = logits[0];
    for (int e = 1; e < kE; ++e) mx = fmaxf(mx, logits[e]);
    float p[kE]; float s = 0.f;
    for (int e = 0; e < kE; ++e) { p[e] = expf(logits[e] - mx); s += p[e]; }
    for (int e = 0; e < kE; ++e) p[e] /= s;
    int i1 = 0;
    for (int e = 1; e < kE; ++e) if (p[e] > p[i1]) i1 = e;
    int i2 = (i1 == 0) ? 1 : 0;
    for (int e = 0; e < kE; ++e) if (e != i1 && p[e] > p[i2]) i2 = e;
    for (int e = 0; e < kE; ++e) wcomb[e] = 0.f;
    wcomb[i1] = p[i1]; wcomb[i2] = p[i2];
  }
  __syncthreads();

  float acc0 = 0.f, acc1 = 0.f;
  const int f = tid >> 4, j = tid & 15;
  for (int e = 0; e < kE; ++e) {
    const float we = wcomb[e];            // uniform across the block
    if (we != 0.f) {
      float ga = 0.f, ua = 0.f;
      const float* wg = Wg + ((long)e * kFF + f) * kD;
      const float* wu = Wu + ((long)e * kFF + f) * kD;
      for (int d = j; d < kD; d += 16) { const float tv = tok[d]; ga += tv * wg[d]; ua += tv * wu[d]; }
      for (int o = 8; o; o >>= 1) { ga += __shfl_xor(ga, o, 16); ua += __shfl_xor(ua, o, 16); }
      if (j == 0) hbuf[f] = (ga / (1.f + expf(-ga))) * ua;  // silu(g)*u
    }
    __syncthreads();
    if (we != 0.f) {
      const int c0 = tid, c1 = tid + 256;
      const float* wd0 = Wd + ((long)e * kD + c0) * kFF;
      const float* wd1 = Wd + ((long)e * kD + c1) * kFF;
      float y0 = 0.f, y1 = 0.f;
      for (int ff = 0; ff < kFF; ++ff) { y0 += hbuf[ff] * wd0[ff]; y1 += hbuf[ff] * wd1[ff]; }
      acc0 += we * (gam[e] * y0 + bet[(long)e * kD + c0]);
      acc1 += we * (gam[e] * y1 + bet[(long)e * kD + c1]);
    }
    __syncthreads();
  }
  moe_out[(long)t * kD + tid]       = acc0;
  moe_out[(long)t * kD + tid + 256] = acc1;
}

// final RevIN denorm
__global__ void denorm_kernel(const float* __restrict__ pin, const float* __restrict__ rev_w,
                              const float* __restrict__ rev_b, const float* __restrict__ meanB,
                              const float* __restrict__ stdB, float* __restrict__ out) {
  long i = (long)blockIdx.x * 256 + threadIdx.x;
  if (i >= (long)kT * kD) return;
  const long bn = i >> 9;
  const int  n  = (int)(bn & (kNV - 1));
  out[i] = (pin[i] - rev_b[n]) / (rev_w[n] + 1e-5f * 1e-5f) * stdB[bn] + meanB[bn];
}

// ---------------------------------------------------------------------------
extern "C" void kernel_launch(void* const* d_in, const int* in_sizes, int n_in,
                              void* d_out, int out_size, void* d_ws, size_t ws_size,
                              hipStream_t stream) {
  (void)in_sizes; (void)n_in; (void)out_size; (void)ws_size;
  const float* x_seq    = (const float*)d_in[0];
  const float* Ins_tk   = (const float*)d_in[1];
  const float* W_emb    = (const float*)d_in[2];
  const float* b_emb    = (const float*)d_in[3];
  const float* rev_w    = (const float*)d_in[4];
  const float* rev_b    = (const float*)d_in[5];
  const float* Wq       = (const float*)d_in[6];
  const float* bq       = (const float*)d_in[7];
  const float* Wk       = (const float*)d_in[8];
  const float* bk       = (const float*)d_in[9];
  const float* Wv       = (const float*)d_in[10];
  const float* bv       = (const float*)d_in[11];
  const float* Wo       = (const float*)d_in[12];
  const float* bo       = (const float*)d_in[13];
  const float* ln1_g    = (const float*)d_in[14];
  const float* ln1_b    = (const float*)d_in[15];
  const float* ln2_g    = (const float*)d_in[16];
  const float* ln2_b    = (const float*)d_in[17];
  const float* gate_W   = (const float*)d_in[18];
  const float* exp_gate = (const float*)d_in[19];
  const float* exp_up   = (const float*)d_in[20];
  const float* exp_down = (const float*)d_in[21];
  const float* eg_w     = (const float*)d_in[22];
  const float* eb_w     = (const float*)d_in[23];
  const float* enc_g    = (const float*)d_in[24];
  const float* enc_b    = (const float*)d_in[25];
  const float* proj_W   = (const float*)d_in[26];
  const float* proj_b   = (const float*)d_in[27];
  float* out = (float*)d_out;

  char* w = (char*)d_ws;
  auto carve = [&](size_t bytes) { void* p = (void*)w; w += (bytes + 255) & ~(size_t)255; return p; };

  float*  meanB   = (float*)carve((size_t)kT * 4);
  float*  stdB    = (float*)carve((size_t)kT * 4);
  float*  insMean = (float*)carve((size_t)kD * 4);
  float*  gamF    = (float*)carve((size_t)kE * 4);
  float*  betF    = (float*)carve((size_t)kE * kD * 4);
  bf16_t* WembBF  = (bf16_t*)carve((size_t)kD * kSL * 2);
  bf16_t* WqBF    = (bf16_t*)carve((size_t)kD * kD * 2);
  bf16_t* WkBF    = (bf16_t*)carve((size_t)kD * kD * 2);
  bf16_t* WvBF    = (bf16_t*)carve((size_t)kD * kD * 2);
  bf16_t* WoBF    = (bf16_t*)carve((size_t)kD * kD * 2);
  bf16_t* WpBF    = (bf16_t*)carve((size_t)kD * kD * 2);
  float*  encF    = (float*)carve((size_t)kT * kD * 4);
  float*  buf1F   = (float*)carve((size_t)kT * kD * 4);
  float*  buf2F   = (float*)carve((size_t)kT * kD * 4);
  float*  buf3F   = (float*)carve((size_t)kT * kD * 4);
  void*   scxn    = carve((size_t)kT * kSL * 2);   // 16MB: xn (bf16) then scores (f32)
  bf16_t* xnBF    = (bf16_t*)scxn;
  float*  scoresF = (float*)scxn;
  bf16_t* pBF     = (bf16_t*)carve((size_t)kH * kNV * kNV * 2);
  bf16_t* encBF   = (bf16_t*)carve((size_t)kT * kD * 2);
  bf16_t* qBF     = (bf16_t*)carve((size_t)kT * kD * 2);
  bf16_t* kBF     = (bf16_t*)carve((size_t)kT * kD * 2);
  bf16_t* oBF     = (bf16_t*)carve((size_t)kT * kD * 2);
  bf16_t* vTBF    = (bf16_t*)carve((size_t)kB * kD * kNV * 2);

  const float scale = 0.17677669529663687f;  // 1/sqrt(dh)

  revin_kernel<<<kT, 256, 0, stream>>>(x_seq, rev_w, rev_b, meanB, stdB, xnBF);
  cast_bf16_kernel<<<(unsigned)(((long)kD * kSL + 255) / 256), 256, 0, stream>>>(W_emb, WembBF, (long)kD * kSL);
  ins_mean_kernel<<<(kD + 255) / 256, 256, 0, stream>>>(Ins_tk, insMean);

  // embedding: enc = xn @ W_emb^T + b_emb   [T,SL]x[D,SL]^T
  wmma_gemm_kernel<<<dim3(kD / BN, kT / BM, 1), 256, 0, stream>>>(
      xnBF, 0, kSL, WembBF, 0, kSL, encF, 0, kD, encBF, 0, b_emb, 1.f, kT, kD, kSL);

  for (int l = 0; l < kNL; ++l) {
    const long wOff = (long)l * kD * kD, bOff = (long)l * kD;
    const unsigned castG = (unsigned)(((long)kD * kD + 255) / 256);
    cast_bf16_kernel<<<castG, 256, 0, stream>>>(Wq + wOff, WqBF, (long)kD * kD);
    cast_bf16_kernel<<<castG, 256, 0, stream>>>(Wk + wOff, WkBF, (long)kD * kD);
    cast_bf16_kernel<<<castG, 256, 0, stream>>>(Wv + wOff, WvBF, (long)kD * kD);
    cast_bf16_kernel<<<castG, 256, 0, stream>>>(Wo + wOff, WoBF, (long)kD * kD);
    eilm_kernel<<<(kE * kD + 255) / 256, 256, 0, stream>>>(
        insMean, eg_w + (long)l * kE * kD, eb_w + (long)l * kE * kD * kD, gamF, betF);

    // Q/K/V projections
    wmma_gemm_kernel<<<dim3(kD / BN, kT / BM, 1), 256, 0, stream>>>(
        encBF, 0, kD, WqBF, 0, kD, buf1F, 0, kD, qBF, 0, bq + bOff, 1.f, kT, kD, kD);
    wmma_gemm_kernel<<<dim3(kD / BN, kT / BM, 1), 256, 0, stream>>>(
        encBF, 0, kD, WkBF, 0, kD, buf2F, 0, kD, kBF, 0, bk + bOff, 1.f, kT, kD, kD);
    wmma_gemm_kernel<<<dim3(kD / BN, kT / BM, 1), 256, 0, stream>>>(
        encBF, 0, kD, WvBF, 0, kD, buf3F, 0, kD, (bf16_t*)nullptr, 0, bv + bOff, 1.f, kT, kD, kD);
    transpose_v_kernel<<<(unsigned)(((long)kT * kD + 255) / 256), 256, 0, stream>>>(buf3F, vTBF);

    for (int b = 0; b < kB; ++b) {
      const long qoff = (long)b * kNV * kD;
      // scores[h] = scale * q_h @ k_h^T  (heads batched on grid.z)
      wmma_gemm_kernel<<<dim3(kNV / BN, kNV / BM, kH), 256, 0, stream>>>(
          qBF + qoff, kDH, kD, kBF + qoff, kDH, kD, scoresF, (long)kNV * kNV, kNV,
          (bf16_t*)nullptr, 0, nullptr, scale, kNV, kNV, kDH);
      softmax_kernel<<<kH * kNV, 256, 0, stream>>>(scoresF, pBF);
      // o_h = P_h @ V_h   (B = V^T rows [dh, NV])
      wmma_gemm_kernel<<<dim3(1, kNV / BM, kH), 256, 0, stream>>>(
          pBF, (long)kNV * kNV, kNV, vTBF + (long)b * kD * kNV, (long)kDH * kNV, kNV,
          buf3F + qoff, kDH, kD, oBF + qoff, kDH, nullptr, 1.f, kNV, kDH, kNV);
    }

    // attention output projection, residual + LN1
    wmma_gemm_kernel<<<dim3(kD / BN, kT / BM, 1), 256, 0, stream>>>(
        oBF, 0, kD, WoBF, 0, kD, buf1F, 0, kD, (bf16_t*)nullptr, 0, bo + bOff, 1.f, kT, kD, kD);
    add_ln_kernel<<<kT, 256, 0, stream>>>(encF, buf1F, ln1_g + bOff, ln1_b + bOff, buf2F, (bf16_t*)nullptr);

    // MoE + EiLM, residual + LN2 -> enc
    moe_kernel<<<kT, 256, 0, stream>>>(buf2F, gate_W + (long)l * kE * kD,
        exp_gate + (long)l * kE * kFF * kD, exp_up + (long)l * kE * kFF * kD,
        exp_down + (long)l * kE * kD * kFF, gamF, betF, buf1F);
    add_ln_kernel<<<kT, 256, 0, stream>>>(buf2F, buf1F, ln2_g + bOff, ln2_b + bOff, encF, encBF);
  }

  // final LN, projection, RevIN denorm
  add_ln_kernel<<<kT, 256, 0, stream>>>(encF, nullptr, enc_g, enc_b, buf2F, qBF);
  cast_bf16_kernel<<<(unsigned)(((long)kD * kD + 255) / 256), 256, 0, stream>>>(proj_W, WpBF, (long)kD * kD);
  wmma_gemm_kernel<<<dim3(kD / BN, kT / BM, 1), 256, 0, stream>>>(
      qBF, 0, kD, WpBF, 0, kD, buf1F, 0, kD, (bf16_t*)nullptr, 0, proj_b, 1.f, kT, kD, kD);
  denorm_kernel<<<(unsigned)(((long)kT * kD + 255) / 256), 256, 0, stream>>>(
      buf1F, rev_w, rev_b, meanB, stdB, out);
}
